// CrissCrossAttention_32023276159245
// MI455X (gfx1250) — matrix operand order repaired and checked by reading, not compiled, not run on hardware
//
#include <hip/hip_runtime.h>
#include <hip/hip_bf16.h>
#include <math.h>

// CrissCrossAttention for MI455X (gfx1250, wave32, WMMA bf16 path).
// B=8, C=256, CQK=32, H=W=128. Scratch needed ~208 MiB.
// B-operand tiles staged in LDS in WMMA fragment order (2x ds_load_b128 per
// fragment); bulk 16B-contiguous copies use GLOBAL_LOAD_ASYNC_TO_LDS_B128
// (ASYNCcnt) when the toolchain exposes the builtin.

#define B_   8
#define C_   256
#define CQK_ 32
#define H_   128
#define W_   128
#define NPIX (H_ * W_)          // 16384
#define NEGINF (-__builtin_inff())

// Fragment-order B-tile storage: 32(K) x 16(N) tile, one 48-byte slot per lane.
#define FR_STR  24              // halfs per lane slot (48 B, 16B-aligned)
#define FR_TILE (32 * FR_STR)   // halfs per tile (1536 B)

#define QT_STR 40               // scores A-tile stride (80 B, 16B-aligned)
#define VT_STR 136              // apply A-tile stride (272 B, 16B-aligned)

typedef __attribute__((ext_vector_type(16))) __bf16 v16bf;
typedef __attribute__((ext_vector_type(8)))  float  v8f;

// ---- CDNA5 async global->LDS (tracked by ASYNCcnt) -----------------------
#define AS1 __attribute__((address_space(1)))
#define AS3 __attribute__((address_space(3)))
typedef int v4i_ __attribute__((vector_size(16)));   // matches builtin proto

#if defined(__has_builtin)
#if __has_builtin(__builtin_amdgcn_global_load_async_to_lds_b128)
#define HAVE_ASYNC_B128 1
#endif
#if __has_builtin(__builtin_amdgcn_s_wait_asynccnt)
#define HAVE_WAIT_ASYNC 1
#endif
#endif
#ifndef HAVE_ASYNC_B128
#define HAVE_ASYNC_B128 0
#endif
#ifndef HAVE_WAIT_ASYNC
#define HAVE_WAIT_ASYNC 0
#endif

__device__ __forceinline__ void async_copy16(void* lds, const void* g) {
#if HAVE_ASYNC_B128
  // (global int4* src, lds int4* dst, imm offset, imm cpol)
  __builtin_amdgcn_global_load_async_to_lds_b128(
      (AS1 v4i_*)(__UINTPTR_TYPE__)g, (AS3 v4i_*)(__UINTPTR_TYPE__)lds, 0, 0);
#else
  *(uint4*)lds = *(const uint4*)g;
#endif
}

__device__ __forceinline__ void wait_async() {
#if HAVE_ASYNC_B128
#if HAVE_WAIT_ASYNC
  __builtin_amdgcn_s_wait_asynccnt(0);
#else
  asm volatile("s_wait_asynccnt 0x0" ::: "memory");
#endif
#endif
}

__device__ __forceinline__ int laneid() { return threadIdx.x & 31; }

// A-matrix 16(M) x 32(K), 16-bit, from row-major LDS tile (stride in halfs).
__device__ __forceinline__ v16bf load_a_frag(const __bf16* tile, int stride) {
  const int l = laneid();
  const __bf16* row = tile + (l & 15) * stride;
  const int kb = (l >> 4) << 4;             // lanes 0-15: K base 0; 16-31: 16
  v16bf a;
#pragma unroll
  for (int j = 0; j < 8; ++j) a[j] = row[kb + j];
#pragma unroll
  for (int j = 0; j < 8; ++j) a[8 + j] = row[kb + 8 + j];
  return a;
}

// B-matrix 32(K) x 16(N) from fragment-order LDS tile: 16 contiguous halfs.
__device__ __forceinline__ v16bf load_b_frag_pk(const __bf16* tile) {
  const __bf16* p = tile + laneid() * FR_STR;
  v16bf b;
#pragma unroll
  for (int j = 0; j < 16; ++j) b[j] = p[j];
  return b;
}

// Fragment-order LDS offset (in halfs) for element (k, n) of a 32x16 B-tile.
__device__ __forceinline__ int bfrag_off(int k, int n) {
  return (((k >> 4) << 4) | n) * FR_STR + (k & 15);
}

// ---------------- Kernel 1: projection GEMM  Out = W*X + bias -------------
// X: (B, C, NPIX) f32   W: (M, C) f32   Out: (B, M, NPIX) bf16
__global__ __launch_bounds__(256) void proj_gemm_kernel(
    const float* __restrict__ X, const float* __restrict__ Wm,
    const float* __restrict__ bias, __bf16* __restrict__ Out, int M) {
  extern __shared__ char smem[];
  __bf16* Wt = (__bf16*)smem;                      // 32 x 34 (A-layout)
  __bf16* Xt = (__bf16*)(smem + 32 * 34 * 2);      // 8 frag-order tiles
  const int n0 = blockIdx.x * 128;
  const int m0 = blockIdx.y * 32;
  const int b  = blockIdx.z;
  const int t  = threadIdx.x;
  const int wave = t >> 5;
  const int mt = wave & 1;                          // 2 M-tiles of 16
  const int ng = wave >> 1;                         // 4 N-groups of 32
  const size_t xb = (size_t)b * C_ * NPIX;
  v8f acc0 = {}, acc1 = {};
  for (int k0 = 0; k0 < C_; k0 += 32) {
    __syncthreads();
#pragma unroll
    for (int i = 0; i < 4; ++i) {                   // 32x32 W tile (A-layout)
      int idx = i * 256 + t;
      int kk = idx & 31, mm = idx >> 5;
      Wt[mm * 34 + kk] = (__bf16)Wm[(size_t)(m0 + mm) * C_ + k0 + kk];
    }
#pragma unroll
    for (int i = 0; i < 4; ++i) {                   // 32x128 X tile, float4 ld
      int chunk = i * 256 + t;                      // 1024 chunks of 4
      int nn = (chunk & 31) * 4;
      int kk = chunk >> 5;
      const float4 xv =
          *(const float4*)&X[xb + (size_t)(k0 + kk) * NPIX + n0 + nn];
      __bf16* dst = Xt + (nn >> 4) * FR_TILE;
      dst[bfrag_off(kk, (nn & 15) + 0)] = (__bf16)xv.x;
      dst[bfrag_off(kk, (nn & 15) + 1)] = (__bf16)xv.y;
      dst[bfrag_off(kk, (nn & 15) + 2)] = (__bf16)xv.z;
      dst[bfrag_off(kk, (nn & 15) + 3)] = (__bf16)xv.w;
    }
    if (k0 + 32 < C_) {                             // prefetch next K tile
      int nn = (t & 31) * 4, kk = t >> 5;
      __builtin_prefetch(&X[xb + (size_t)(k0 + 32 + kk) * NPIX + n0 + nn], 0, 3);
    }
    __syncthreads();
    v16bf a  = load_a_frag(Wt + mt * 16 * 34, 34);
    v16bf b0 = load_b_frag_pk(Xt + (ng * 2 + 0) * FR_TILE);
    v16bf b1 = load_b_frag_pk(Xt + (ng * 2 + 1) * FR_TILE);
    acc0 = __builtin_amdgcn_wmma_f32_16x16x32_bf16(false, a, false, b0,
                                                   (short)0, acc0, false, false);
    acc1 = __builtin_amdgcn_wmma_f32_16x16x32_bf16(false, a, false, b1,
                                                   (short)0, acc1, false, false);
  }
  const int l = laneid();
  const int mrow = mt * 16 + ((l >> 4) << 3);
  const int ncol = n0 + ng * 32 + (l & 15);
  const size_t ob = (size_t)b * M * NPIX;
#pragma unroll
  for (int j = 0; j < 8; ++j) {
    int m = m0 + mrow + j;
    float bv = bias[m];
    Out[ob + (size_t)m * NPIX + ncol]      = (__bf16)(acc0[j] + bv);
    Out[ob + (size_t)m * NPIX + ncol + 16] = (__bf16)(acc1[j] + bv);
  }
}

// ---------------- Kernel 2: attention scores S = Q^T K (128x128) ----------
// mode 0: per (b,w) column -> eH (diag masked); mode 1: per (b,h) row -> eW
__global__ __launch_bounds__(256) void scores_kernel(
    const __bf16* __restrict__ q, const __bf16* __restrict__ k,
    float* __restrict__ out, int mode) {
  extern __shared__ char smem[];
  __bf16* Qt = (__bf16*)smem;                       // 128 x QT_STR (A-layout)
  __bf16* Kt = (__bf16*)(smem + 128 * QT_STR * 2);  // 8 frag-order tiles
  const int s = blockIdx.x;                         // w (mode0) or h (mode1)
  const int b = blockIdx.y;
  const int t = threadIdx.x;
  const size_t base = (size_t)b * CQK_ * NPIX +
                      (mode == 0 ? (size_t)s : (size_t)s * W_);
  const int mStride = (mode == 0) ? W_ : 1;
  {
    // Q^T staging: thread owns one m row-half; gather 16 strided halfs,
    // write 2x b128 to LDS.
    const int m = t & 127, ch = t >> 7;
    union { __bf16 h[16]; uint4 u[2]; } g;
#pragma unroll
    for (int j = 0; j < 16; ++j)
      g.h[j] = q[base + (size_t)(ch * 16 + j) * NPIX + (size_t)m * mStride];
    uint4* dst = (uint4*)(Qt + m * QT_STR + ch * 16);
    dst[0] = g.u[0];
    dst[1] = g.u[1];
  }
  {
    // K staging straight into fragment order: thread owns one lane slot
    // (n, k-half); gather 16 strided halfs, write 2x b128.
    const int n = t & 127, kh = t >> 7;
    union { __bf16 h[16]; uint4 u[2]; } g;
#pragma unroll
    for (int j = 0; j < 16; ++j)
      g.h[j] = k[base + (size_t)(kh * 16 + j) * NPIX + (size_t)n * mStride];
    __bf16* slot =
        Kt + (n >> 4) * FR_TILE + (((kh << 4) | (n & 15)) * FR_STR);
    ((uint4*)slot)[0] = g.u[0];
    *(uint4*)(slot + 8) = g.u[1];
  }
  __syncthreads();
  const int wave = t >> 5;
  const int l = laneid();
  v16bf a = load_a_frag(Qt + wave * 16 * QT_STR, QT_STR);
  size_t outBase, rowStride;
  if (mode == 0) { outBase = (size_t)b * H_ * W_ * H_ + (size_t)s * H_;
                   rowStride = (size_t)W_ * H_; }
  else           { outBase = (size_t)b * H_ * W_ * W_ + (size_t)s * W_ * W_;
                   rowStride = W_; }
  const int mrow = wave * 16 + ((l >> 4) << 3);
  const int nc0 = l & 15;
  v8f zero = {};
#pragma unroll
  for (int tn = 0; tn < 8; ++tn) {
    v16bf bf = load_b_frag_pk(Kt + tn * FR_TILE);
    v8f acc = __builtin_amdgcn_wmma_f32_16x16x32_bf16(false, a, false, bf,
                                                      (short)0, zero, false, false);
    int n = tn * 16 + nc0;
#pragma unroll
    for (int j = 0; j < 8; ++j) {
      int m = mrow + j;
      float sc = acc[j];
      if (mode == 0 && m == n) sc = NEGINF;         // mask diagonal of eH
      out[outBase + (size_t)m * rowStride + n] = sc;
    }
  }
}

// ---------------- Kernel 3: exact top-64 mask + joint softmax -------------
__device__ __forceinline__ unsigned fkey(float f) {   // monotone float->uint
  unsigned u = __float_as_uint(f);
  return (u & 0x80000000u) ? ~u : (u | 0x80000000u);
}

__global__ __launch_bounds__(256) void topk_softmax_kernel(
    const float* __restrict__ eH, const float* __restrict__ eW,
    __bf16* __restrict__ attH, __bf16* __restrict__ attW) {
  const int wave = threadIdx.x >> 5;
  const int l = threadIdx.x & 31;
  const size_t pix = (size_t)blockIdx.x * 8 + wave;  // (b,h,w) linear
  const float4 vH = ((const float4*)(eH + pix * 128))[l];
  const float4 vW = ((const float4*)(eW + pix * 128))[l];
  float aH[4] = {vH.x, vH.y, vH.z, vH.w};
  float aW[4] = {vW.x, vW.y, vW.z, vW.w};
  unsigned kH[4], kW[4];
#pragma unroll
  for (int j = 0; j < 4; ++j) { kH[j] = fkey(aH[j]); kW[j] = fkey(aW[j]); }
  unsigned thrH = 0, thrW = 0;                       // greedy MSB bit search
#pragma unroll
  for (int bit = 31; bit >= 0; --bit) {
    unsigned pH = thrH | (1u << bit), pW = thrW | (1u << bit);
    int c = 0;
#pragma unroll
    for (int j = 0; j < 4; ++j) {
      c += (kH[j] >= pH) ? 1 : 0;
      c += (kW[j] >= pW) ? (1 << 16) : 0;
    }
#pragma unroll
    for (int off = 16; off > 0; off >>= 1) c += __shfl_xor(c, off, 32);
    if ((c & 0xFFFF) >= 64) thrH = pH;
    if ((c >> 16)    >= 64) thrW = pW;
  }
  float m = -__builtin_inff();
#pragma unroll
  for (int j = 0; j < 4; ++j) {
    if (kH[j] >= thrH) m = fmaxf(m, aH[j]);
    if (kW[j] >= thrW) m = fmaxf(m, aW[j]);
  }
#pragma unroll
  for (int off = 16; off > 0; off >>= 1) m = fmaxf(m, __shfl_xor(m, off, 32));
  float s = 0.f, pHv[4], pWv[4];
#pragma unroll
  for (int j = 0; j < 4; ++j) {
    pHv[j] = (kH[j] >= thrH) ? __expf(aH[j] - m) : 0.f;
    pWv[j] = (kW[j] >= thrW) ? __expf(aW[j] - m) : 0.f;
    s += pHv[j] + pWv[j];
  }
#pragma unroll
  for (int off = 16; off > 0; off >>= 1) s += __shfl_xor(s, off, 32);
  float inv = 1.0f / s;
  union { __bf16 h[4]; uint2 u2; } oH, oW;
#pragma unroll
  for (int j = 0; j < 4; ++j) {
    oH.h[j] = (__bf16)(pHv[j] * inv);
    oW.h[j] = (__bf16)(pWv[j] * inv);
  }
  // att rows (bf16, stride 256) alias the fp32 score rows in place:
  // this wave already has its own 512B row in registers.
  ((uint2*)(attH + pix * 256))[l] = oH.u2;
  ((uint2*)(attW + pix * 256))[l] = oW.u2;
}

// -------- Shared apply-GEMM body: Out(c,n) += V(c,k) x Att^T(k,n) ---------
// Vt: 256 x VT_STR A-layout; Bt: 32 frag-order tiles (4 kblk x 8 ntile).
__device__ __forceinline__ void apply_gemm(
    const __bf16* Vt, const __bf16* Bt, v8f (&res)[8][2]) {
  const int wave = threadIdx.x >> 5;
  const int cBase = wave * 32;
  v16bf afr[2][4];
#pragma unroll
  for (int ct = 0; ct < 2; ++ct)
#pragma unroll
    for (int ks = 0; ks < 4; ++ks)
      afr[ct][ks] =
          load_a_frag(Vt + (cBase + ct * 16) * VT_STR + ks * 32, VT_STR);
  v8f zero = {};
#pragma unroll
  for (int nt = 0; nt < 8; ++nt) {
    v16bf bfr[4];
#pragma unroll
    for (int ks = 0; ks < 4; ++ks)
      bfr[ks] = load_b_frag_pk(Bt + (ks * 8 + nt) * FR_TILE);
#pragma unroll
    for (int ct = 0; ct < 2; ++ct) {
      v8f acc = zero;
#pragma unroll
      for (int ks = 0; ks < 4; ++ks)
        acc = __builtin_amdgcn_wmma_f32_16x16x32_bf16(false, afr[ct][ks], false,
                                                      bfr[ks], (short)0, acc,
                                                      false, false);
      res[nt][ct] = acc;
    }
  }
}

// ---------------- Kernel 4: outH = V_col x attH^T  (writes d_out) ---------
__global__ __launch_bounds__(256) void applyH_kernel(
    const __bf16* __restrict__ v, const __bf16* __restrict__ attH,
    float* __restrict__ out) {
  extern __shared__ char smem[];
  __bf16* Vt = (__bf16*)smem;                       // 256 x VT_STR (c x g)
  __bf16* Bt = (__bf16*)(smem + 256 * VT_STR * 2);  // frag-order att tiles
  const int w = blockIdx.x, b = blockIdx.y;
  const int t = threadIdx.x;
  const size_t vb = (size_t)b * C_ * NPIX;
#pragma unroll
  for (int i = 0; i < 16; ++i) {  // Vt[c][g]=v[b][c][g][w]: gather 8, b128 st
    int chunk = i * 256 + t;      // 4096 chunks of 8 g
    int g0 = (chunk & 15) * 8;
    int c  = chunk >> 4;
    union { __bf16 h[8]; uint4 u; } tmp;
#pragma unroll
    for (int j = 0; j < 8; ++j)
      tmp.h[j] = v[vb + (size_t)c * NPIX + (size_t)(g0 + j) * W_ + w];
    *(uint4*)(Vt + c * VT_STR + g0) = tmp.u;
  }
#pragma unroll
  for (int i = 0; i < 8; ++i) {   // att -> frag tiles: async 16B copies
    int chunk = i * 256 + t;      // 2048 chunks of 8 g
    int g0 = (chunk & 15) * 8;
    int hh = chunk >> 4;
    async_copy16(Bt + (g0 >> 5) * 8 * FR_TILE + (hh >> 4) * FR_TILE +
                     bfrag_off(g0 & 31, hh & 15),
                 &attH[(((size_t)b * H_ + hh) * W_ + w) * 256 + g0]);
  }
  wait_async();
  __syncthreads();
  v8f res[8][2];
  apply_gemm(Vt, Bt, res);
  const int l = t & 31;
  const int cBase = (t >> 5) * 32 + ((l >> 4) << 3);
  const int ncol = l & 15;
#pragma unroll
  for (int nt = 0; nt < 8; ++nt) {
    int h0 = nt * 16 + ncol;
#pragma unroll
    for (int ct = 0; ct < 2; ++ct)
#pragma unroll
      for (int j = 0; j < 8; ++j) {
        int c = cBase + ct * 16 + j;
        out[vb + (size_t)c * NPIX + (size_t)h0 * W_ + w] = res[nt][ct][j];
      }
  }
}

// ---- Kernel 5: outW = V_row x attW^T; d_out = gamma*(outH+outW)+x1 -------
__global__ __launch_bounds__(256) void applyW_kernel(
    const __bf16* __restrict__ v, const __bf16* __restrict__ attW,
    const float* __restrict__ x1, const float* __restrict__ gammaP,
    float* __restrict__ out) {
  extern __shared__ char smem[];
  __bf16* Vt = (__bf16*)smem;                       // 256 x VT_STR (c x v)
  __bf16* Bt = (__bf16*)(smem + 256 * VT_STR * 2);  // frag-order att tiles
  const int h = blockIdx.x, b = blockIdx.y;
  const int t = threadIdx.x;
  const float gamma = gammaP[0];
  const size_t vb = (size_t)b * C_ * NPIX;
  const size_t rowb = vb + (size_t)h * W_;
#pragma unroll
  for (int i = 0; i < 16; ++i) {  // Vt rows are contiguous: async 16B copies
    int chunk = i * 256 + t;      // 4096 chunks of 8 vv
    int v0 = (chunk & 15) * 8;
    int c  = chunk >> 4;
    async_copy16(Vt + c * VT_STR + v0, &v[rowb + (size_t)c * NPIX + v0]);
  }
#pragma unroll
  for (int i = 0; i < 8; ++i) {   // att -> frag tiles: async 16B copies
    int chunk = i * 256 + t;
    int v0 = (chunk & 15) * 8;
    int ww = chunk >> 4;
    async_copy16(Bt + (v0 >> 5) * 8 * FR_TILE + (ww >> 4) * FR_TILE +
                     bfrag_off(v0 & 31, ww & 15),
                 &attW[(((size_t)b * H_ + h) * W_ + ww) * 256 + v0]);
  }
  wait_async();
  __syncthreads();
  v8f res[8][2];
  apply_gemm(Vt, Bt, res);
  const int l = t & 31;
  const int cBase = (t >> 5) * 32 + ((l >> 4) << 3);
  const int ncol = l & 15;
#pragma unroll
  for (int nt = 0; nt < 8; ++nt) {
    int w0 = nt * 16 + ncol;
#pragma unroll
    for (int ct = 0; ct < 2; ++ct)
#pragma unroll
      for (int j = 0; j < 8; ++j) {
        int c = cBase + ct * 16 + j;
        size_t addr = rowb + (size_t)c * NPIX + w0;
        out[addr] = gamma * (out[addr] + res[nt][ct][j]) + x1[addr];
      }
  }
}

// ---------------------------------------------------------------------------
extern "C" void kernel_launch(void* const* d_in, const int* in_sizes, int n_in,
                              void* d_out, int out_size, void* d_ws, size_t ws_size,
                              hipStream_t stream) {
  const float* x1 = (const float*)d_in[0];
  const float* x2 = (const float*)d_in[1];
  const float* Wq = (const float*)d_in[2];
  const float* bq = (const float*)d_in[3];
  const float* Wk = (const float*)d_in[4];
  const float* bk = (const float*)d_in[5];
  const float* Wv = (const float*)d_in[6];
  const float* bv = (const float*)d_in[7];
  const float* gm = (const float*)d_in[8];
  float* out = (float*)d_out;

  char* ws = (char*)d_ws;
  size_t off = 0;
  __bf16* qb = (__bf16*)(ws + off); off += (size_t)B_ * CQK_ * NPIX * 2;
  __bf16* kb = (__bf16*)(ws + off); off += (size_t)B_ * CQK_ * NPIX * 2;
  __bf16* vb = (__bf16*)(ws + off); off += (size_t)B_ * C_ * NPIX * 2;
  float*  eH = (float*)(ws + off);  off += (size_t)B_ * NPIX * H_ * 4;
  float*  eW = (float*)(ws + off);  off += (size_t)B_ * NPIX * W_ * 4;
  if (off > ws_size) return;                      // needs ~208 MiB scratch
  __bf16* attH = (__bf16*)eH;                     // bf16 rows alias f32 rows
  __bf16* attW = (__bf16*)eW;

  const size_t smem1 = 32 * 34 * 2 + 8 * FR_TILE * 2;
  proj_gemm_kernel<<<dim3(NPIX / 128, CQK_ / 32, B_), 256, smem1, stream>>>(
      x1, Wq, bq, qb, CQK_);
  proj_gemm_kernel<<<dim3(NPIX / 128, CQK_ / 32, B_), 256, smem1, stream>>>(
      x2, Wk, bk, kb, CQK_);
  proj_gemm_kernel<<<dim3(NPIX / 128, C_ / 32, B_), 256, smem1, stream>>>(
      x2, Wv, bv, vb, C_);

  const size_t smem2 = 128 * QT_STR * 2 + 8 * FR_TILE * 2;
  scores_kernel<<<dim3(W_, B_), 256, smem2, stream>>>(qb, kb, eH, 0);
  scores_kernel<<<dim3(H_, B_), 256, smem2, stream>>>(qb, kb, eW, 1);

  topk_softmax_kernel<<<dim3(B_ * NPIX / 8), 256, 0, stream>>>(eH, eW, attH, attW);

  const size_t smem4 = 256 * VT_STR * 2 + 32 * FR_TILE * 2;  // ~118 KB
  applyH_kernel<<<dim3(W_, B_), 256, smem4, stream>>>(vb, attH, out);
  applyW_kernel<<<dim3(H_, B_), 256, smem4, stream>>>(vb, attW, x1, gm, out);
}
